// AdaptiveThresholdSNN_4913442587074
// MI455X (gfx1250) — matrix operand order repaired and checked by reading, not compiled, not run on hardware
//
#include <hip/hip_runtime.h>
#include <hip/hip_bf16.h>

typedef __attribute__((ext_vector_type(8)))  float  v8f;
typedef __attribute__((ext_vector_type(2)))  float  v2f;
typedef __attribute__((ext_vector_type(16))) __bf16 v16bf;
typedef __attribute__((ext_vector_type(8)))  __bf16 v8bf;

#define SNN_BETA 0.95f

// ---------------------------------------------------------------------------
// Kernel 1: x1[t*B+b][h] = sum_i x[t*B+b][i] * W1[h][i] + b1[h]
// M = 25600, N = 800, K = 784.  One 16x16 tile per wave, f32 WMMA (exact).
// A layout (f32 16x4): lane<16: M=lane, K={k,k+1}; lane>=16: M=lane-16, K={k+2,k+3}
// ---------------------------------------------------------------------------
__global__ __launch_bounds__(256) void snn_proj(const float* __restrict__ x,
                                                const float* __restrict__ W1,
                                                const float* __restrict__ b1,
                                                float* __restrict__ x1) {
    const int lane = threadIdx.x & 31;
    const int warp = threadIdx.x >> 5;
    const int g    = blockIdx.x * 8 + warp;      // tile id, 0..79999
    const int tm   = g / 50;
    const int tn   = g - tm * 50;
    const int r0   = tm * 16;
    const int h0   = tn * 16;
    const int lm   = lane & 15;
    const int lhi  = lane >> 4;

    const float* arow = x  + (size_t)(r0 + lm) * 784 + 2 * lhi;
    const float* brow = W1 + (size_t)(h0 + lm) * 784 + 2 * lhi;

    v8f c = {};
    for (int k = 0; k < 784; k += 4) {
        v2f a = *(const v2f*)(arow + k);
        v2f b = *(const v2f*)(brow + k);
        c = __builtin_amdgcn_wmma_f32_16x16x4_f32(false, a, false, b,
                                                  (short)0, c, false, false);
    }
    const float bias = b1[h0 + lm];
    float* orow = x1 + (size_t)(r0 + 8 * lhi) * 800 + h0 + lm;
#pragma unroll
    for (int r = 0; r < 8; ++r) orow[(size_t)r * 800] = c[r] + bias;
}

// ---------------------------------------------------------------------------
// Kernel 2: split W2 into 3 bf16 planes (hi/mid/lo) for fp32-accurate bf16 WMMA
// ---------------------------------------------------------------------------
__global__ __launch_bounds__(256) void snn_split_w2(const float* __restrict__ W2,
                                                    __bf16* __restrict__ wh,
                                                    __bf16* __restrict__ wm,
                                                    __bf16* __restrict__ wl,
                                                    int n) {
    int i = blockIdx.x * 256 + threadIdx.x;
    if (i >= n) return;
    float w   = W2[i];
    __bf16 a  = (__bf16)w;
    float r1  = w - (float)a;
    __bf16 b  = (__bf16)r1;
    float r2  = r1 - (float)b;
    __bf16 cc = (__bf16)r2;
    wh[i] = a; wm[i] = b; wl[i] = cc;
}

// ---------------------------------------------------------------------------
// Kernel 3: persistent fused scan. 32 blocks x 256 threads; block owns 16 batch
// rows for all 50 timesteps. Membrane state in registers, spikes via LDS.
// ---------------------------------------------------------------------------
__global__ __launch_bounds__(256) void snn_scan(
    const float*  __restrict__ x1,
    const __bf16* __restrict__ W2h, const __bf16* __restrict__ W2m,
    const __bf16* __restrict__ W2l,
    const float*  __restrict__ b2,  const float* __restrict__ W3,
    const float*  __restrict__ b3,
    const float*  __restrict__ thr1, const float* __restrict__ thr2,
    float* __restrict__ out_sum, float* __restrict__ s1r,
    float* __restrict__ s2r,     float* __restrict__ s3r) {

    __shared__ __bf16 spk1_lds[16 * 808];
    __shared__ __bf16 spk2_lds[16 * 808];

    const int tid  = threadIdx.x;
    const int lane = tid & 31;
    const int warp = tid >> 5;
    const int b0   = blockIdx.x * 16;
    const int lm   = lane & 15;
    const int lhi  = lane >> 4;

    // layer-1 membrane state: thread owns elements tid + 256*j of the 16x800 tile
    float m1[50];
#pragma unroll
    for (int j = 0; j < 50; ++j) m1[j] = 0.f;

    // layer-2 membrane state lives in the WMMA accumulator mapping:
    // tile ti -> columns (warp+8*ti)*16 .. +15 ; element (r, lm) -> (b0+r+8*lhi, n0+lm)
    v8f m2[7];
#pragma unroll
    for (int ti = 0; ti < 7; ++ti) m2[ti] = (v8f){};

    float m3 = 0.f, osum = 0.f;
    const int bl3 = tid / 10;
    const int o3  = tid - bl3 * 10;
    float b3v = 0.f;
    if (tid < 160) b3v = b3[o3];

    for (int t = 0; t < 50; ++t) {
        // ---------------- phase A: LIF layer 1 (elementwise) ----------------
        const float* x1t = x1  + (size_t)t * 409600 + (size_t)b0 * 800;
        float*       s1t = s1r + (size_t)t * 409600 + (size_t)b0 * 800;
#pragma unroll
        for (int j = 0; j < 50; ++j) {
            int idx = tid + 256 * j;
            int bl  = idx / 800;
            int h   = idx - bl * 800;
            float v   = SNN_BETA * m1[j] + x1t[idx];
            float spk = (v > thr1[h]) ? 1.f : 0.f;
            m1[j] = v * (1.f - spk);
            s1t[idx] = spk;
            spk1_lds[bl * 808 + h] = (__bf16)spk;
        }
        __syncthreads();

        // ------- phase B: mem2 = beta*mem2 + spk1 @ W2^T + b2, spike -------
        float* s2t = s2r + (size_t)t * 409600 + (size_t)b0 * 800;
        const __bf16* arow = spk1_lds + lm * 808;   // A row M = lane&15
#pragma unroll
        for (int ti = 0; ti < 7; ++ti) {
            const int tile = warp + 8 * ti;
            if (tile < 50) {
                const int n0   = tile * 16;
                const int hcol = n0 + lm;
                const float b2v = b2[hcol];
                const float t2v = thr2[hcol];
                v8f c = m2[ti];
#pragma unroll
                for (int r = 0; r < 8; ++r) c[r] = c[r] * SNN_BETA + b2v;

                const __bf16* rh = W2h + (size_t)hcol * 800;
                const __bf16* rm = W2m + (size_t)hcol * 800;
                const __bf16* rl = W2l + (size_t)hcol * 800;

                for (int kc = 0; kc < 800; kc += 32) {
                    // A frag: lane<16 K kc..kc+15; lane>=16 K kc+16..kc+31 (contig)
                    v8bf alo = *(const v8bf*)(arow + kc + 16 * lhi);
                    v8bf ahi = *(const v8bf*)(arow + kc + 16 * lhi + 8);
                    v16bf a  = __builtin_shufflevector(alo, ahi,
                        0,1,2,3,4,5,6,7,8,9,10,11,12,13,14,15);
                    // B frag: lane<16 holds K kc..kc+15 of column hcol (contig)
                    v16bf bh = *(const v16bf*)(rh + kc + 16 * lhi);
                    v16bf bm = *(const v16bf*)(rm + kc + 16 * lhi);
                    v16bf bl = *(const v16bf*)(rl + kc + 16 * lhi);
                    c = __builtin_amdgcn_wmma_f32_16x16x32_bf16(false, a, false, bh,
                                                                (short)0, c, false, false);
                    c = __builtin_amdgcn_wmma_f32_16x16x32_bf16(false, a, false, bm,
                                                                (short)0, c, false, false);
                    c = __builtin_amdgcn_wmma_f32_16x16x32_bf16(false, a, false, bl,
                                                                (short)0, c, false, false);
                }
#pragma unroll
                for (int r = 0; r < 8; ++r) {
                    float v   = c[r];
                    float spk = (v > t2v) ? 1.f : 0.f;
                    v *= (1.f - spk);
                    c[r] = v;
                    int bl_ = r + 8 * lhi;
                    s2t[(size_t)bl_ * 800 + hcol] = spk;
                    spk2_lds[bl_ * 808 + hcol] = (__bf16)spk;
                }
                m2[ti] = c;
            }
        }
        __syncthreads();

        // ------- phase C: mem3 = beta*mem3 + spk2 @ W3^T + b3, spike -------
        float* s3t = s3r + (size_t)t * 5120 + b0 * 10;
        if (tid < 160) {
            float acc3 = m3 * SNN_BETA + b3v;
            const float*  w3row = W3 + (size_t)o3 * 800;
            const __bf16* srow  = spk2_lds + bl3 * 808;
            for (int k = 0; k < 800; k += 8) {
                v8bf s8 = *(const v8bf*)(srow + k);
                v8f  w8 = *(const v8f*)(w3row + k);
#pragma unroll
                for (int q = 0; q < 8; ++q) acc3 += (float)s8[q] * w8[q];
            }
            float spk = (acc3 > 1.0f) ? 1.f : 0.f;
            m3 = acc3 * (1.f - spk);
            s3t[bl3 * 10 + o3] = spk;
            osum += spk;
        }
        __syncthreads();
    }

    if (tid < 160) out_sum[(size_t)(b0 + bl3) * 10 + o3] = osum;
}

// ---------------------------------------------------------------------------
extern "C" void kernel_launch(void* const* d_in, const int* in_sizes, int n_in,
                              void* d_out, int out_size, void* d_ws, size_t ws_size,
                              hipStream_t stream) {
    const float* x    = (const float*)d_in[0];
    const float* W1   = (const float*)d_in[1];
    const float* b1   = (const float*)d_in[2];
    const float* W2   = (const float*)d_in[3];
    const float* b2   = (const float*)d_in[4];
    const float* W3   = (const float*)d_in[5];
    const float* b3   = (const float*)d_in[6];
    const float* thr1 = (const float*)d_in[7];
    const float* thr2 = (const float*)d_in[8];

    // workspace layout
    float*  x1  = (float*)d_ws;                                 // 20,480,000 f32
    __bf16* W2h = (__bf16*)((char*)d_ws + 81920000);            // 640,000 bf16
    __bf16* W2m = W2h + 640000;
    __bf16* W2l = W2m + 640000;

    // output layout: [sum 512*10][s1r 50*512*800][s2r 50*512*800][s3r 50*512*10]
    float* out_sum = (float*)d_out;
    float* s1r = out_sum + 5120;
    float* s2r = s1r + 20480000;
    float* s3r = s2r + 20480000;

    snn_proj<<<10000, 256, 0, stream>>>(x, W1, b1, x1);
    snn_split_w2<<<2500, 256, 0, stream>>>(W2, W2h, W2m, W2l, 640000);
    snn_scan<<<32, 256, 0, stream>>>(x1, W2h, W2m, W2l, b2, W3, b3,
                                     thr1, thr2, out_sum, s1r, s2r, s3r);
}
